// MemTransformerBlock_5592047420070
// MI455X (gfx1250) — compile-verified, matrix operand order checked by reading, not run on hardware
//
#include <hip/hip_runtime.h>

// ---------------------------------------------------------------------------
// MemTransformerBlock for MI455X (gfx1250, wave32, WMMA).
// bf16 WMMA (v_wmma_f32_16x16x32_bf16) with f32 accumulate.
// Async global->LDS (ASYNCcnt) double-buffered GEMM pipeline when available.
// ---------------------------------------------------------------------------

typedef __attribute__((ext_vector_type(8)))  unsigned short v8u;
typedef __attribute__((ext_vector_type(16))) unsigned short v16u;
typedef __attribute__((ext_vector_type(16))) __bf16        v16bf;
typedef __attribute__((ext_vector_type(8)))  float         v8f;
typedef __attribute__((ext_vector_type(4)))  int           v4i;

#define S_LEN   2048L
#define H_DIM   2048L
#define HD_DIM  128L
#define NQ_H    16L
#define NKV_H   4L
#define NMEM    64L
#define H4_DIM  8192L
#define KV_DIM  512L

#if __has_builtin(__builtin_amdgcn_global_load_async_to_lds_b128) && \
    __has_builtin(__builtin_amdgcn_s_wait_asynccnt)
#define USE_ASYNC_LDS 1
typedef __attribute__((address_space(1))) v4i* gbl_v4i_ptr;
typedef __attribute__((address_space(3))) v4i* lds_v4i_ptr;
#else
#define USE_ASYNC_LDS 0
#endif

__device__ __forceinline__ unsigned short f2bf(float f) {
  unsigned int u = __float_as_uint(f);
  u += 0x7FFFu + ((u >> 16) & 1u);          // round-to-nearest-even
  return (unsigned short)(u >> 16);
}

union FragU { v16u u; v16bf b; };

// Stage one 128x32 bf16 tile (row-major, K-contiguous) into an LDS buffer
// (row stride 40 elems). 256 threads -> 2 x b128 per thread.
__device__ __forceinline__ void stage_tile(
    const unsigned short* __restrict__ G, long ld, long base0, int maxIdx,
    int kk, unsigned short* lds, int tid)
{
  #pragma unroll
  for (int c = tid; c < 512; c += 256) {
    int row = c >> 2;
    int col = (c & 3) << 3;
    long gr = base0 + row; if (gr >= maxIdx) gr = maxIdx - 1;
    const unsigned short* gp = G + gr * ld + (kk + col);
#if USE_ASYNC_LDS
    unsigned short* gpm = const_cast<unsigned short*>(gp);
    __builtin_amdgcn_global_load_async_to_lds_b128(
        (gbl_v4i_ptr)(v4i*)gpm,
        (lds_v4i_ptr)(v4i*)(lds + row * 40 + col),
        0, 0);
#else
    *(v8u*)(lds + row * 40 + col) = *(const v8u*)gp;
#endif
    __builtin_prefetch((const void*)(gp + 32), 0, 1);   // next K tile
  }
}

// ---------------------------------------------------------------------------
// bf16 GEMM:  C[m,n] = epilogue( scale * sum_k A[m,k] * B[n,k] )
//   A: bf16 row-major, leading dim lda (K contiguous).
//   B: bf16 row-major, leading dim ldb (K contiguous; "row" n = output col).
//   batch via blockIdx.z: A += zA*z, C += zC*z.
//   epilogue: mask (col > row+maskOff -> -1e30), residual add (resPtr, ldc),
//             SiLU, f32 store (Cf), bf16 store (Cb).
// Tile: 128x128 per workgroup, 8 waves each 64x32, K-step 32,
// double-buffered LDS, one barrier per K-step.
// ---------------------------------------------------------------------------
__global__ __launch_bounds__(256) void gemm_bf16_kernel(
    const unsigned short* __restrict__ A, long lda, long zA,
    const unsigned short* __restrict__ B, long ldb,
    float* __restrict__ Cf, unsigned short* __restrict__ Cb, long ldc, long zC,
    const float* __restrict__ resPtr,
    int M, int N, int K, float scale, int maskOff, int act)
{
  __shared__ __align__(16) unsigned short As[2][128 * 40];
  __shared__ __align__(16) unsigned short Bs[2][128 * 40];

  const int tid   = threadIdx.x;
  const int lane  = tid & 31;
  const int wave  = tid >> 5;
  const int half  = lane >> 4;
  const int r     = lane & 15;
  const int waveM = wave >> 2;   // 0..1
  const int waveN = wave & 3;    // 0..3

  A += zA * blockIdx.z;
  const long m0 = (long)blockIdx.y * 128;
  const long n0 = (long)blockIdx.x * 128;

  v8f acc[4][2];
  #pragma unroll
  for (int mi = 0; mi < 4; ++mi)
    #pragma unroll
    for (int ni = 0; ni < 2; ++ni)
      #pragma unroll
      for (int e = 0; e < 8; ++e) acc[mi][ni][e] = 0.0f;

  const int nt = K >> 5;               // K / 32, K always a multiple of 32
  stage_tile(A, lda, m0, M, 0, As[0], tid);
  stage_tile(B, ldb, n0, N, 0, Bs[0], tid);

  for (int it = 0; it < nt; ++it) {
    const int cur = it & 1;
#if USE_ASYNC_LDS
    __builtin_amdgcn_s_wait_asynccnt(0);   // my tile-it DMAs complete
#endif
    __syncthreads();                       // everyone's tile-it data in LDS;
                                           // also: all reads of buf[cur^1] done
    if (it + 1 < nt) {                     // stream next tile during compute
      stage_tile(A, lda, m0, M, (it + 1) << 5, As[cur ^ 1], tid);
      stage_tile(B, ldb, n0, N, (it + 1) << 5, Bs[cur ^ 1], tid);
    }

    // Fragments per ISA 16-bit A/B layout:
    // lane r holds row r, K chunks [half*8, +8) and [16+half*8, +8).
    FragU afr[4], bfr[2];
    #pragma unroll
    for (int mi = 0; mi < 4; ++mi) {
      const unsigned short* p = &As[cur][(waveM * 64 + mi * 16 + r) * 40 + half * 8];
      v8u lo = *(const v8u*)(p);
      v8u hi = *(const v8u*)(p + 16);
      afr[mi].u = __builtin_shufflevector(lo, hi, 0,1,2,3,4,5,6,7,8,9,10,11,12,13,14,15);
    }
    #pragma unroll
    for (int ni = 0; ni < 2; ++ni) {
      const unsigned short* p = &Bs[cur][(waveN * 32 + ni * 16 + r) * 40 + half * 8];
      v8u lo = *(const v8u*)(p);
      v8u hi = *(const v8u*)(p + 16);
      bfr[ni].u = __builtin_shufflevector(lo, hi, 0,1,2,3,4,5,6,7,8,9,10,11,12,13,14,15);
    }
    #pragma unroll
    for (int mi = 0; mi < 4; ++mi)
      #pragma unroll
      for (int ni = 0; ni < 2; ++ni)
        acc[mi][ni] = __builtin_amdgcn_wmma_f32_16x16x32_bf16(
            false, afr[mi].b, false, bfr[ni].b, (short)0, acc[mi][ni], false, false);
  }

  if (Cf) Cf += zC * blockIdx.z;
  if (Cb) Cb += zC * blockIdx.z;

  // C layout: VGPR v, lane l -> row = v + 8*(l>>4), col = l & 15
  #pragma unroll
  for (int mi = 0; mi < 4; ++mi) {
    #pragma unroll
    for (int ni = 0; ni < 2; ++ni) {
      #pragma unroll
      for (int v = 0; v < 8; ++v) {
        long row = m0 + waveM * 64 + mi * 16 + v + 8 * half;
        long col = n0 + waveN * 32 + ni * 16 + r;
        if (row < M && col < N) {
          float val = acc[mi][ni][v] * scale;
          if (maskOff >= 0 && col > row + maskOff) val = -1e30f;
          if (resPtr) val += resPtr[row * ldc + col];
          if (act) val = val / (1.0f + __expf(-val));   // SiLU
          if (Cf) Cf[row * ldc + col] = val;
          if (Cb) Cb[row * ldc + col] = f2bf(val);
        }
      }
    }
  }
}

// ---------------------------------------------------------------------------
// x = l0*x + l1*residual   (f32 out + bf16 copy)
// ---------------------------------------------------------------------------
__global__ __launch_bounds__(256) void mix_kernel(
    const float* __restrict__ x, const float* __restrict__ res,
    const float* __restrict__ lam, float* __restrict__ xf,
    unsigned short* __restrict__ xb, long n)
{
  long i = (long)blockIdx.x * 256 + threadIdx.x;
  if (i < n) {
    float v = lam[0] * x[i] + lam[1] * res[i];
    xf[i] = v;
    xb[i] = f2bf(v);
  }
}

__global__ __launch_bounds__(256) void f32_to_bf16_kernel(
    const float* __restrict__ a, unsigned short* __restrict__ o, long n)
{
  long i = (long)blockIdx.x * 256 + threadIdx.x;
  if (i < n) o[i] = f2bf(a[i]);
}

// ---------------------------------------------------------------------------
// vt[(hk*128+d)*S + m] = vb[m*512 + hk*128 + d]  (bf16 transpose of V)
// Run over `rows` leading tokens.
// ---------------------------------------------------------------------------
__global__ __launch_bounds__(256) void vtranspose_kernel(
    const unsigned short* __restrict__ vb, unsigned short* __restrict__ vt,
    long rows)
{
  long i = (long)blockIdx.x * 256 + threadIdx.x;
  if (i < rows * KV_DIM) {
    long m = i >> 9;        // token
    long c = i & 511;       // hk*128 + d
    vt[c * S_LEN + m] = vb[i];
  }
}

// ---------------------------------------------------------------------------
// Per-(token, head) RMSNorm over HD=128 then RoPE; write bf16.
// grid = (tokens, heads), block = 128
// ---------------------------------------------------------------------------
__global__ __launch_bounds__(128) void rmsrope_kernel(
    const float* __restrict__ X, unsigned short* __restrict__ Y,
    int heads, int pos0)
{
  __shared__ float red[128];
  const int t = blockIdx.x;
  const int h = blockIdx.y;
  const long base = ((long)t * heads + h) * 128;
  const int tid = threadIdx.x;
  float v = X[base + tid];
  red[tid] = v * v; __syncthreads();
  for (int st = 64; st > 0; st >>= 1) {
    if (tid < st) red[tid] += red[tid + st];
    __syncthreads();
  }
  float inv = rsqrtf(red[0] * (1.0f / 128.0f) + 1e-5f);
  if (tid < 64) {
    float x1 = X[base + tid] * inv;
    float x2 = X[base + tid + 64] * inv;
    // inv_freq = 10000^(-2d/128) = exp(-(2d/128)*ln(10000))
    float fr = (float)(pos0 + t) *
               __expf(-((float)(2 * tid) * (1.0f / 128.0f)) * 9.210340371976184f);
    float c = cosf(fr), s = sinf(fr);
    Y[base + tid]      = f2bf(x1 * c + x2 * s);
    Y[base + tid + 64] = f2bf(-x1 * s + x2 * c);
  }
}

// ---------------------------------------------------------------------------
// Per-row RMSNorm over H=2048, write bf16. grid = rows, block = 256
// ---------------------------------------------------------------------------
__global__ __launch_bounds__(256) void rmsnorm_row_kernel(
    const float* __restrict__ X, unsigned short* __restrict__ Y)
{
  __shared__ float red[256];
  const long row = blockIdx.x;
  const float* x = X + row * 2048;
  const int tid = threadIdx.x;
  float vals[8]; float ss = 0.0f;
  #pragma unroll
  for (int i = 0; i < 8; ++i) { float v = x[tid + i * 256]; vals[i] = v; ss += v * v; }
  red[tid] = ss; __syncthreads();
  for (int st = 128; st > 0; st >>= 1) {
    if (tid < st) red[tid] += red[tid + st];
    __syncthreads();
  }
  float inv = rsqrtf(red[0] * (1.0f / 2048.0f) + 1e-5f);
  #pragma unroll
  for (int i = 0; i < 8; ++i) Y[row * 2048 + tid + i * 256] = f2bf(vals[i] * inv);
}

// ---------------------------------------------------------------------------
// Row softmax over L=2048 f32 scores -> bf16 probs. grid = rows, block = 256
// ---------------------------------------------------------------------------
__global__ __launch_bounds__(256) void softmax2048_kernel(
    const float* __restrict__ Sc, unsigned short* __restrict__ P)
{
  __shared__ float red[256];
  const long row = blockIdx.x;
  const float* s = Sc + row * 2048;
  unsigned short* p = P + row * 2048;
  const int tid = threadIdx.x;
  float vals[8];
  float mx = -3.4e38f;
  #pragma unroll
  for (int i = 0; i < 8; ++i) { float v = s[tid + i * 256]; vals[i] = v; mx = fmaxf(mx, v); }
  red[tid] = mx; __syncthreads();
  for (int st = 128; st > 0; st >>= 1) {
    if (tid < st) red[tid] = fmaxf(red[tid], red[tid + st]);
    __syncthreads();
  }
  mx = red[0]; __syncthreads();
  float sum = 0.0f;
  #pragma unroll
  for (int i = 0; i < 8; ++i) { float e = __expf(vals[i] - mx); vals[i] = e; sum += e; }
  red[tid] = sum; __syncthreads();
  for (int st = 128; st > 0; st >>= 1) {
    if (tid < st) red[tid] += red[tid + st];
    __syncthreads();
  }
  float inv = 1.0f / red[0];
  #pragma unroll
  for (int i = 0; i < 8; ++i) p[tid + i * 256] = f2bf(vals[i] * inv);
}

// ---------------------------------------------------------------------------
// Host-side orchestration
// ---------------------------------------------------------------------------
static inline void gemm_go(hipStream_t st,
    const unsigned short* A, long lda, long zA,
    const unsigned short* B, long ldb,
    float* Cf, unsigned short* Cb, long ldc, long zC,
    const float* res, long M, long N, long K,
    float scale, int maskOff, int act, int batch)
{
  dim3 grid((unsigned)((N + 127) / 128), (unsigned)((M + 127) / 128), (unsigned)batch);
  gemm_bf16_kernel<<<grid, dim3(256), 0, st>>>(A, lda, zA, B, ldb,
                                               Cf, Cb, ldc, zC, res,
                                               (int)M, (int)N, (int)K,
                                               scale, maskOff, act);
}

extern "C" void kernel_launch(void* const* d_in, const int* in_sizes, int n_in,
                              void* d_out, int out_size, void* d_ws, size_t ws_size,
                              hipStream_t stream)
{
  (void)in_sizes; (void)n_in; (void)out_size; (void)ws_size;
  const float* x_in    = (const float*)d_in[0];
  const float* resid   = (const float*)d_in[3];
  const float* Wq      = (const float*)d_in[6];
  const float* Wk      = (const float*)d_in[7];
  const float* Wv      = (const float*)d_in[8];
  const float* Wkm     = (const float*)d_in[9];
  const float* Wvm     = (const float*)d_in[10];
  const float* Wo      = (const float*)d_in[11];
  const float* Wh      = (const float*)d_in[12];
  const float* Wm      = (const float*)d_in[13];
  const float* lambdas = (const float*)d_in[14];
  float* out = (float*)d_out;

  char* ws = (char*)d_ws;
  size_t off = 0;
  auto alloc = [&](size_t bytes) -> void* {
    void* p = ws + off; off += (bytes + 255) & ~(size_t)255; return p;
  };
  unsigned short* wqb   = (unsigned short*)alloc(H_DIM * H_DIM * 2);
  unsigned short* wkb   = (unsigned short*)alloc(KV_DIM * H_DIM * 2);
  unsigned short* wvb   = (unsigned short*)alloc(KV_DIM * H_DIM * 2);
  unsigned short* wkmb  = (unsigned short*)alloc(KV_DIM * H_DIM * 2);
  unsigned short* wvmb  = (unsigned short*)alloc(KV_DIM * H_DIM * 2);
  unsigned short* wob   = (unsigned short*)alloc(H_DIM * H_DIM * 2);
  unsigned short* whb   = (unsigned short*)alloc(H4_DIM * H_DIM * 2);
  unsigned short* wmb   = (unsigned short*)alloc(H_DIM * H4_DIM * 2);
  float*          xf    = (float*)alloc(S_LEN * H_DIM * 4);
  unsigned short* xb    = (unsigned short*)alloc(S_LEN * H_DIM * 2);
  float*          kf    = (float*)alloc(S_LEN * KV_DIM * 4);
  unsigned short* kb    = (unsigned short*)alloc(S_LEN * KV_DIM * 2);
  unsigned short* vb    = (unsigned short*)alloc(S_LEN * KV_DIM * 2);
  unsigned short* vt    = (unsigned short*)alloc(S_LEN * KV_DIM * 2);  // V^T
  float*          qf    = (float*)alloc(S_LEN * H_DIM * 4);
  unsigned short* qb    = (unsigned short*)alloc(S_LEN * H_DIM * 2);
  float*          sbuf  = (float*)alloc(4 * S_LEN * S_LEN * 4);
  unsigned short* pb    = (unsigned short*)alloc(4 * S_LEN * S_LEN * 2);
  unsigned short* attnb = (unsigned short*)alloc(S_LEN * H_DIM * 2);
  float*          x1f   = (float*)alloc(S_LEN * H_DIM * 4);
  unsigned short* x1nb  = (unsigned short*)alloc(S_LEN * H_DIM * 2);
  unsigned short* hb    = (unsigned short*)alloc(S_LEN * H4_DIM * 2);
  unsigned short* xoutb = (unsigned short*)alloc(NMEM * H_DIM * 2);

  auto conv = [&](const float* src, unsigned short* dst, long n) {
    f32_to_bf16_kernel<<<dim3((unsigned)((n + 255) / 256)), dim3(256), 0, stream>>>(src, dst, n);
  };
  conv(Wq,  wqb,  H_DIM * H_DIM);
  conv(Wk,  wkb,  KV_DIM * H_DIM);
  conv(Wv,  wvb,  KV_DIM * H_DIM);
  conv(Wkm, wkmb, KV_DIM * H_DIM);
  conv(Wvm, wvmb, KV_DIM * H_DIM);
  conv(Wo,  wob,  H_DIM * H_DIM);
  conv(Wh,  whb,  H4_DIM * H_DIM);
  conv(Wm,  wmb,  H_DIM * H4_DIM);

  mix_kernel<<<dim3((unsigned)((S_LEN * H_DIM + 255) / 256)), dim3(256), 0, stream>>>(
      x_in, resid, lambdas, xf, xb, S_LEN * H_DIM);

  // K/V projections over full sequence (k -> rms+rope -> kb ; v -> vb -> vt)
  gemm_go(stream, xb, H_DIM, 0, wkb, H_DIM, kf, nullptr, KV_DIM, 0, nullptr,
          S_LEN, KV_DIM, H_DIM, 1.0f, -1, 0, 1);
  rmsrope_kernel<<<dim3((unsigned)S_LEN, (unsigned)NKV_H), dim3(128), 0, stream>>>(
      kf, kb, (int)NKV_H, 0);
  gemm_go(stream, xb, H_DIM, 0, wvb, H_DIM, nullptr, vb, KV_DIM, 0, nullptr,
          S_LEN, KV_DIM, H_DIM, 1.0f, -1, 0, 1);
  vtranspose_kernel<<<dim3((unsigned)((S_LEN * KV_DIM + 255) / 256)), dim3(256), 0, stream>>>(
      vb, vt, S_LEN);

  auto run_block = [&](long row0, long n, int maskOff, unsigned short* out_bf) {
    // q projection + per-head rms/rope
    gemm_go(stream, xb + row0 * H_DIM, H_DIM, 0, wqb, H_DIM,
            qf, nullptr, H_DIM, 0, nullptr, n, H_DIM, H_DIM, 1.0f, -1, 0, 1);
    rmsrope_kernel<<<dim3((unsigned)n, (unsigned)NQ_H), dim3(128), 0, stream>>>(
        qf, qb, (int)NQ_H, (int)row0);
    // attention per kv head, g=4 query heads batched in grid.z
    for (int hk = 0; hk < 4; ++hk) {
      gemm_go(stream, qb + hk * 4 * HD_DIM, H_DIM, HD_DIM,
              kb + hk * HD_DIM, KV_DIM,
              sbuf, nullptr, S_LEN, n * S_LEN, nullptr,
              n, S_LEN, HD_DIM, 0.08838834764831845f, maskOff, 0, 4);
      softmax2048_kernel<<<dim3((unsigned)(4 * n)), dim3(256), 0, stream>>>(sbuf, pb);
      gemm_go(stream, pb, S_LEN, n * S_LEN,
              vt + hk * HD_DIM * S_LEN, S_LEN,
              nullptr, attnb + hk * 4 * HD_DIM, H_DIM, HD_DIM, nullptr,
              n, HD_DIM, S_LEN, 1.0f, -1, 0, 4);
    }
    // x1 = x + attn @ Wo.T
    gemm_go(stream, attnb, H_DIM, 0, wob, H_DIM,
            x1f, nullptr, H_DIM, 0, xf + row0 * H_DIM,
            n, H_DIM, H_DIM, 1.0f, -1, 0, 1);
    // h = silu(rms(x1) @ Wh.T)
    rmsnorm_row_kernel<<<dim3((unsigned)n), dim3(256), 0, stream>>>(x1f, x1nb);
    gemm_go(stream, x1nb, H_DIM, 0, whb, H_DIM,
            nullptr, hb, H4_DIM, 0, nullptr,
            n, H4_DIM, H_DIM, 1.0f, -1, 1, 1);
    // out = x1 + h @ Wm.T
    gemm_go(stream, hb, H4_DIM, 0, wmb, H4_DIM,
            out + row0 * H_DIM, out_bf, H_DIM, 0, x1f,
            n, H_DIM, H4_DIM, 1.0f, -1, 0, 1);
  };

  // memory block: all-ones mask
  run_block(0, NMEM, -1, xoutb);

  // k_mem / v_mem re-projection from x_mem (overwrite first NMEM rows)
  gemm_go(stream, xoutb, H_DIM, 0, wkmb, H_DIM, kf, nullptr, KV_DIM, 0, nullptr,
          NMEM, KV_DIM, H_DIM, 1.0f, -1, 0, 1);
  rmsrope_kernel<<<dim3((unsigned)NMEM, (unsigned)NKV_H), dim3(128), 0, stream>>>(
      kf, kb, (int)NKV_H, 0);
  gemm_go(stream, xoutb, H_DIM, 0, wvmb, H_DIM, nullptr, vb, KV_DIM, 0, nullptr,
          NMEM, KV_DIM, H_DIM, 1.0f, -1, 0, 1);
  vtranspose_kernel<<<dim3((unsigned)((NMEM * KV_DIM + 255) / 256)), dim3(256), 0, stream>>>(
      vb, vt, NMEM);

  // main block: allowed iff m <= n_local + NMEM  (memory + causal)
  run_block(NMEM, S_LEN - NMEM, (int)NMEM, nullptr);
}
